// PatchedMatmul_10333691314383
// MI455X (gfx1250) — 8-bit matrix operand order repaired and checked by reading, compiled, not run on hardware
//
#include <hip/hip_runtime.h>


typedef __attribute__((ext_vector_type(16))) int   v16i;
typedef __attribute__((ext_vector_type(8)))  float v8f;
typedef int v4i_vs __attribute__((vector_size(16)));

static constexpr int M = 16384;
static constexpr int N = 4096;
static constexpr int K = 4096;
static constexpr float F8_MAX = 448.0f;

#if __has_builtin(__builtin_amdgcn_global_load_async_to_lds_b128)
#define HAVE_ASYNC_LDS 1
#else
#define HAVE_ASYNC_LDS 0
#endif

#if __has_builtin(__builtin_amdgcn_s_wait_asynccnt)
#define WAIT_ASYNC(n) __builtin_amdgcn_s_wait_asynccnt(n)
#else
#define WAIT_ASYNC(n) asm volatile("s_wait_asynccnt %0" ::"i"(n) : "memory")
#endif

#define AS_GLOBAL(p) ((__attribute__((address_space(1))) v4i_vs*)(p))
#define AS_LDS(p)    ((__attribute__((address_space(3))) v4i_vs*)(p))


__device__ __forceinline__ unsigned char sw_fp8(float f) {
  unsigned u = __float_as_uint(f);
  unsigned s = (u >> 31) << 7;
  int exp8 = (int)((u >> 23) & 0xff);
  unsigned m = u & 0x7fffff;
  if (exp8 == 0) return (unsigned char)s;
  int E = (exp8 - 127) + 7;
  if (E >= 1) {
    unsigned t = m + 0x7ffff + ((m >> 20) & 1);
    unsigned man = t >> 20;
    if (man >= 8) { man = 0; ++E; }
    if (E > 15) { E = 15; man = 6; }
    if (E == 15 && man == 7) man = 6;
    return (unsigned char)(s | ((unsigned)E << 4) | man);
  }
  unsigned full = 0x800000u | m;
  int sh = 20 + (1 - E);
  if (sh > 24) return (unsigned char)s;
  unsigned man = full >> sh;
  unsigned rem = full & ((1u << sh) - 1u);
  unsigned half = 1u << (sh - 1);
  if (rem > half || (rem == half && (man & 1))) ++man;
  if (man >= 8) return (unsigned char)(s | (1u << 4));
  return (unsigned char)(s | man);
}

__device__ __forceinline__ unsigned pack4_fp8(float a, float b, float c, float d) {
#if __has_builtin(__builtin_amdgcn_cvt_pk_fp8_f32)
  int p = __builtin_amdgcn_cvt_pk_fp8_f32(a, b, 0, false);
  p = __builtin_amdgcn_cvt_pk_fp8_f32(c, d, p, true);
  return (unsigned)p;
#else
  return (unsigned)sw_fp8(a) | ((unsigned)sw_fp8(b) << 8) |
         ((unsigned)sw_fp8(c) << 16) | ((unsigned)sw_fp8(d) << 24);
#endif
}

__device__ __forceinline__ unsigned char fp8_byte(float a) {
#if __has_builtin(__builtin_amdgcn_cvt_pk_fp8_f32)
  return (unsigned char)(__builtin_amdgcn_cvt_pk_fp8_f32(a, a, 0, false) & 0xff);
#else
  return sw_fp8(a);
#endif
}

__device__ __forceinline__ float clamp_q(float x, float inv) {
  return fminf(fmaxf(x * inv, -F8_MAX), F8_MAX);
}

__device__ __forceinline__ unsigned short f2bf16(float f) {
  unsigned u = __float_as_uint(f);
  unsigned r = u + 0x7fffu + ((u >> 16) & 1u);
  return (unsigned short)(r >> 16);
}


__global__ __launch_bounds__(256) void quant_x_fp8(
    const float* __restrict__ x, const float* __restrict__ scale,
    unsigned char* __restrict__ xq, long long n4) {
  long long i = (long long)blockIdx.x * 256 + threadIdx.x;
  if (i >= n4) return;
  float inv = 1.0f / scale[0];
  float4 v = ((const float4*)x)[i];
  ((unsigned*)xq)[i] = pack4_fp8(clamp_q(v.x, inv), clamp_q(v.y, inv),
                                 clamp_q(v.z, inv), clamp_q(v.w, inv));
}


__global__ __launch_bounds__(256) void quant_w_fp8_t(
    const float* __restrict__ w, const float* __restrict__ scale,
    unsigned char* __restrict__ wt) {
  __shared__ unsigned char tile[64 * 64];
  float inv = 1.0f / scale[0];
  int k0 = blockIdx.y * 64, n0 = blockIdx.x * 64;
  int tid = threadIdx.x;
  #pragma unroll
  for (int it = 0; it < 16; ++it) {
    int id = tid + it * 256;
    int kk = id >> 6, nn = id & 63;
    float f = w[(size_t)(k0 + kk) * N + (n0 + nn)];
    tile[nn * 64 + kk] = fp8_byte(clamp_q(f, inv));
  }
  __syncthreads();
  int nn = tid >> 2, ko = (tid & 3) * 16;
  *(int4*)(wt + (size_t)(n0 + nn) * K + (k0 + ko)) =
      *(const int4*)(tile + nn * 64 + ko);
}


__global__ __launch_bounds__(256) void gemm_fp8_wmma(
    const unsigned char* __restrict__ Aq,
    const unsigned char* __restrict__ Bt,
    const float* __restrict__ sA, const float* __restrict__ sB,
    unsigned short* __restrict__ out) {
  __shared__ unsigned char As[2][128 * 128];
  __shared__ unsigned char Bs[2][128 * 128];

  const int tid = threadIdx.x;
  const int wave = tid >> 5, lane = tid & 31;
  const int l16 = lane & 15, hi = lane >> 4;
  const int wr = wave & 3, wc = wave >> 2;
  const int bm = blockIdx.y * 128, bn = blockIdx.x * 128;

  v8f acc[2][4];
  #pragma unroll
  for (int s = 0; s < 2; ++s)
    #pragma unroll
    for (int j = 0; j < 4; ++j) acc[s][j] = (v8f)(0.0f);

  auto stage = [&](int buf, int kb) {
    #pragma unroll
    for (int i = 0; i < 4; ++i) {
      int c = tid + i * 256;
      int row = c >> 3, ko = (c & 7) * 16;
      const unsigned char* ga = Aq + (size_t)(bm + row) * K + kb + ko;
      const unsigned char* gb = Bt + (size_t)(bn + row) * K + kb + ko;
#if HAVE_ASYNC_LDS
      __builtin_amdgcn_global_load_async_to_lds_b128(
          AS_GLOBAL(ga), AS_LDS(&As[buf][c * 16]), 0, 0);
      __builtin_amdgcn_global_load_async_to_lds_b128(
          AS_GLOBAL(gb), AS_LDS(&Bs[buf][c * 16]), 0, 0);
#else
      *(int4*)(&As[buf][c * 16]) = *(const int4*)ga;
      *(int4*)(&Bs[buf][c * 16]) = *(const int4*)gb;
#endif
    }
  };

  stage(0, 0);

  for (int kb = 0; kb < K; kb += 128) {
    const int cur = (kb >> 7) & 1;
    const bool more = (kb + 128) < K;
    if (more) stage(cur ^ 1, kb + 128);
#if HAVE_ASYNC_LDS
    if (more) WAIT_ASYNC(8); else WAIT_ASYNC(0);
#endif
    __syncthreads();

    v16i afrag[2];
    #pragma unroll
    for (int s = 0; s < 2; ++s) {
      union { v16i v; unsigned long long q[8]; } ua;
      const unsigned char* ap =
          &As[cur][(wr * 32 + s * 16 + l16) * 128 + hi * 8];
      #pragma unroll
      for (int h = 0; h < 2; ++h)
        #pragma unroll
        for (int cc = 0; cc < 4; ++cc)
          ua.q[h * 4 + cc] =
              *(const unsigned long long*)(ap + h * 64 + cc * 16);
      afrag[s] = ua.v;
    }

    #pragma unroll
    for (int j = 0; j < 4; ++j) {
      union { v16i v; unsigned long long x[8]; } ub;
      const unsigned char* bp =
          &Bs[cur][(wc * 64 + j * 16 + l16) * 128 + hi * 8];
      #pragma unroll
      for (int cc = 0; cc < 8; ++cc)
        ub.x[cc] = *(const unsigned long long*)(bp + (cc >> 2) * 64 + (cc & 3) * 16);
      #pragma unroll
      for (int s = 0; s < 2; ++s)
        acc[s][j] = __builtin_amdgcn_wmma_f32_16x16x128_fp8_fp8(
            afrag[s], ub.v, (short)0, acc[s][j], false, false);
    }
    __syncthreads();
  }

  const float sc = sA[0] * sB[0];
  #pragma unroll
  for (int s = 0; s < 2; ++s) {
    #pragma unroll
    for (int j = 0; j < 4; ++j) {
      int m0 = bm + wr * 32 + s * 16 + hi * 8;
      int n0 = bn + wc * 64 + j * 16 + l16;
      #pragma unroll
      for (int v = 0; v < 8; ++v)
        out[(size_t)(m0 + v) * N + n0] = f2bf16(acc[s][j][v] * sc);
    }
  }
}


extern "C" void kernel_launch(void* const* d_in, const int* in_sizes, int n_in,
                              void* d_out, int out_size, void* d_ws,
                              size_t ws_size, hipStream_t stream) {
  const float* x  = (const float*)d_in[0];
  const float* w  = (const float*)d_in[1];
  const float* si = (const float*)d_in[2];
  const float* so = (const float*)d_in[3];
  unsigned short* out = (unsigned short*)d_out;

  unsigned char* xq = (unsigned char*)d_ws;
  unsigned char* wt = xq + (size_t)M * K;

  long long n4 = (long long)M * K / 4;
  quant_x_fp8<<<(unsigned)((n4 + 255) / 256), 256, 0, stream>>>(x, si, xq, n4);
  quant_w_fp8_t<<<dim3(N / 64, K / 64), 256, 0, stream>>>(w, so, wt);
  gemm_fp8_wmma<<<dim3(N / 128, M / 128), 256, 0, stream>>>(xq, wt, si, so, out);
}
